// AttentionBlock2_14345190768932
// MI455X (gfx1250) — matrix operand order repaired and checked by reading, not compiled, not run on hardware
//
#include <hip/hip_runtime.h>
#include <hip/hip_bf16.h>

typedef __attribute__((ext_vector_type(16))) __bf16 v16bf;
typedef __attribute__((ext_vector_type(8)))  float  v8f;

#define NB   2      // batches
#define CV   64     // v_feat channels
#define CR   20     // r_feat channels
#define EE   64     // embed dim
#define CO   64     // out channels
#define TN   32     // nodes per workgroup
#define COLS 128    // gathered v columns per workgroup (TN * bundle)

union BF16x16 { v16bf v; __bf16 h[16]; };

__device__ __forceinline__ __bf16 f2bf(float f) { return (__bf16)f; }  // native cvt

__device__ __forceinline__ v8f vzero8() {
    v8f z;
#pragma unroll
    for (int i = 0; i < 8; ++i) z[i] = 0.f;
    return z;
}

// A-fragment (16x32 bf16) for one wave, rows [m0,m0+16), K base kbase.
// 16-bit A layout: lane<16 holds K = kbase+[0,8)+[16,24); lane>=16 holds +8.
__device__ __forceinline__ v16bf load_a_global(const float* Wrow_kbase, int hi) {
    BF16x16 r;
    const float* p0 = Wrow_kbase + 8 * hi;
    const float* p1 = p0 + 16;
#pragma unroll
    for (int i = 0; i < 8; ++i) r.h[i]     = f2bf(p0[i]);
#pragma unroll
    for (int i = 0; i < 8; ++i) r.h[8 + i] = f2bf(p1[i]);
    return r.v;
}

// Wq A-fragment with K padded 20 -> 32 (zeros)
__device__ __forceinline__ v16bf load_a_wq(const float* Wq, int row, int hi) {
    BF16x16 r;
#pragma unroll
    for (int i = 0; i < 8; ++i) {
        int c = 8 * hi + i;
        r.h[i] = (c < CR) ? f2bf(Wq[row * CR + c]) : (__bf16)0.0f;
    }
#pragma unroll
    for (int i = 0; i < 8; ++i) {
        int c = 16 + 8 * hi + i;
        r.h[8 + i] = (c < CR) ? f2bf(Wq[row * CR + c]) : (__bf16)0.0f;
    }
    return r.v;
}

__global__ void zero_f32(float* p, int n) {
    int i = blockIdx.x * blockDim.x + threadIdx.x;
    if (i < n) p[i] = 0.f;
}

__global__ __launch_bounds__(128) void attn_block_kernel(
    const float* __restrict__ v_feat, const float* __restrict__ r_feat,
    const float* __restrict__ Wq, const float* __restrict__ Wk,
    const float* __restrict__ Wv, const float* __restrict__ Wo,
    const int* __restrict__ v2p, const int* __restrict__ r2p,
    float* __restrict__ out, int Mv, int Mr, int N, int Nn)
{
    __shared__ __align__(64) __bf16 s_vbuf[CV * COLS]; // gathered v, bf16 [K=c][N=col]
    __shared__ __align__(64) __bf16 s_rbuf[32 * TN];   // gathered r, bf16, K padded to 32
    __shared__ __align__(64) __bf16 s_agg [EE * TN];   // agg bf16 [K=e][N=node]
    __shared__ __align__(16) float s_qbuf[EE * TN];    // q f32 [e][node]
    __shared__ __align__(16) float s_vres[EE * COLS];  // V-projection f32 [e][col]
    __shared__ float s_score[COLS];
    __shared__ float s_attn [COLS];
    __shared__ int   s_vidx [COLS];
    __shared__ int   s_ridx [TN];

    const int tid  = threadIdx.x;
    const int lane = tid & 31;
    const int wv   = tid >> 5;       // wave 0..3 -> E-row stripe
    const int hi   = lane >> 4;
    const int lm   = lane & 15;
    const int m0   = 16 * wv;

    const int tiles_per_b = Nn / TN;
    const int b  = blockIdx.x / tiles_per_b;
    const int n0 = (blockIdx.x % tiles_per_b) * TN;

    // ---- preload A fragments (registers, bf16) ----
    const int arow = m0 + lm;
    v16bf aWq = load_a_wq(Wq, arow, hi);
    v16bf aWk[2], aWv[2], aWo[2];
#pragma unroll
    for (int ks = 0; ks < 2; ++ks) {
        aWk[ks] = load_a_global(Wk + arow * CV + ks * 32, hi);
        aWv[ks] = load_a_global(Wv + arow * CV + ks * 32, hi);
        aWo[ks] = load_a_global(Wo + arow * EE + ks * 32, hi);
    }

    // ---- Phase A: indices + gathered activations -> LDS (bf16) ----
    if (tid < COLS) {
        s_vidx[tid]  = v2p[((size_t)b * N + (size_t)n0 * 4 + tid) * 2];
        s_score[tid] = 0.f;
    }
    if (tid < TN) s_ridx[tid] = r2p[((size_t)b * Nn + n0 + tid) * 2];
    __syncthreads();

    const float* vb = v_feat + (size_t)b * CV * Mv;
    for (int i = tid; i < CV * COLS; i += 128) {
        int c = i >> 7, col = i & (COLS - 1);
        s_vbuf[i] = f2bf(vb[(size_t)c * Mv + s_vidx[col]]);
    }
    const float* rb = r_feat + (size_t)b * CR * Mr;
    for (int i = tid; i < 32 * TN; i += 128) {
        int c = i >> 5, n = i & (TN - 1);
        float val = (c < CR) ? rb[(size_t)c * Mr + s_ridx[n]] : 0.f;
        s_rbuf[i] = f2bf(val);
    }
    __syncthreads();

    // ---- Phase B: q = Wq * r  (WMMA bf16 -> f32) ----
#pragma unroll
    for (int nt = 0; nt < 2; ++nt) {
        BF16x16 ub;
#pragma unroll
        for (int s = 0; s < 16; ++s)
            ub.h[s] = s_rbuf[(8 * hi + (s & 7) + 16 * (s >> 3)) * TN + nt * 16 + lm];
        v16bf bq = ub.v;
        v8f acc = vzero8();
        acc = __builtin_amdgcn_wmma_f32_16x16x32_bf16(false, aWq, false, bq,
                                                      (short)0, acc, false, false);
#pragma unroll
        for (int j = 0; j < 8; ++j)
            s_qbuf[(m0 + j + 8 * hi) * TN + nt * 16 + lm] = acc[j];
    }
    __syncthreads();

    // ---- Phase C: K/V projections + partial scores ----
    for (int nt = 0; nt < 8; ++nt) {
        v8f kacc = vzero8(), vacc = vzero8();
#pragma unroll
        for (int ks = 0; ks < 2; ++ks) {
            BF16x16 ub;
#pragma unroll
            for (int s = 0; s < 16; ++s)
                ub.h[s] = s_vbuf[(ks * 32 + 8 * hi + (s & 7) + 16 * (s >> 3)) * COLS + nt * 16 + lm];
            v16bf bf = ub.v;
            kacc = __builtin_amdgcn_wmma_f32_16x16x32_bf16(false, aWk[ks], false, bf,
                                                           (short)0, kacc, false, false);
            vacc = __builtin_amdgcn_wmma_f32_16x16x32_bf16(false, aWv[ks], false, bf,
                                                           (short)0, vacc, false, false);
        }
        const int col = nt * 16 + lm;
        const int nn  = col >> 2;
#pragma unroll
        for (int j = 0; j < 8; ++j)
            s_vres[(m0 + j + 8 * hi) * COLS + col] = vacc[j];
        float p = 0.f;
#pragma unroll
        for (int j = 0; j < 8; ++j)
            p += s_qbuf[(m0 + j + 8 * hi) * TN + nn] * kacc[j];
        p += __shfl_xor(p, 16, 32);           // sum the wave's 16 E-rows
        if (hi == 0) atomicAdd(&s_score[col], p);  // cross-wave via LDS
    }
    __syncthreads();

    // ---- Phase D: 4-wide softmax + agg ----
    {
        float s  = s_score[tid] * 0.125f;     // 1/sqrt(64)
        float mx = s;
        mx = fmaxf(mx, __shfl_xor(mx, 1, 32));
        mx = fmaxf(mx, __shfl_xor(mx, 2, 32));
        float ex = __expf(s - mx);
        float sm = ex;
        sm += __shfl_xor(sm, 1, 32);
        sm += __shfl_xor(sm, 2, 32);
        s_attn[tid] = ex / sm;
    }
    __syncthreads();

    for (int i = tid; i < EE * TN; i += 128) {
        int e = i >> 5, n = i & (TN - 1);
        const float* vr = &s_vres[e * COLS + n * 4];
        const float* at = &s_attn[n * 4];
        float s = at[0] * vr[0] + at[1] * vr[1] + at[2] * vr[2] + at[3] * vr[3];
        s_agg[e * TN + n] = f2bf(s);
    }
    __syncthreads();

    // ---- Phase E: xr = Wo * agg, scatter-add to output ----
    float* outb = out + (size_t)b * CO * Mr;
#pragma unroll
    for (int nt = 0; nt < 2; ++nt) {
        v8f acc = vzero8();
#pragma unroll
        for (int ks = 0; ks < 2; ++ks) {
            BF16x16 ub;
#pragma unroll
            for (int s = 0; s < 16; ++s)
                ub.h[s] = s_agg[(ks * 32 + 8 * hi + (s & 7) + 16 * (s >> 3)) * TN + nt * 16 + lm];
            v16bf bf = ub.v;
            acc = __builtin_amdgcn_wmma_f32_16x16x32_bf16(false, aWo[ks], false, bf,
                                                          (short)0, acc, false, false);
        }
#pragma unroll
        for (int j = 0; j < 8; ++j) {
            int o = m0 + j + 8 * hi;
            int n = nt * 16 + lm;
            atomicAdd(&outb[(size_t)o * Mr + s_ridx[n]], acc[j]);
        }
    }
}

extern "C" void kernel_launch(void* const* d_in, const int* in_sizes, int n_in,
                              void* d_out, int out_size, void* d_ws, size_t ws_size,
                              hipStream_t stream) {
    (void)n_in; (void)d_ws; (void)ws_size;
    const float* v_feat = (const float*)d_in[0];
    const float* r_feat = (const float*)d_in[1];
    const float* Wq     = (const float*)d_in[2];
    const float* Wk     = (const float*)d_in[3];
    const float* Wv     = (const float*)d_in[4];
    const float* Wo     = (const float*)d_in[5];
    const int*   v2p    = (const int*)d_in[6];
    const int*   r2p    = (const int*)d_in[7];
    // d_in[8] = os (==8 -> bundle 4, baked into the kernel tiling)

    const int Mv = in_sizes[0] / (NB * CV);
    const int Mr = in_sizes[1] / (NB * CR);
    const int N  = in_sizes[6] / (NB * 2);
    const int Nn = in_sizes[7] / (NB * 2);

    float* out = (float*)d_out;
    zero_f32<<<(out_size + 255) / 256, 256, 0, stream>>>(out, out_size);

    const int grid = NB * (Nn / TN);
    attn_block_kernel<<<grid, 128, 0, stream>>>(v_feat, r_feat, Wq, Wk, Wv, Wo,
                                                v2p, r2p, out, Mv, Mr, N, Nn);
}
